// GATLayer_33303176413264
// MI455X (gfx1250) — matrix operand order repaired and checked by reading, not compiled, not run on hardware
//
#include <hip/hip_runtime.h>
#include <hip/hip_bf16.h>
#include <math.h>

typedef __fp16   v2hf __attribute__((ext_vector_type(2)));
typedef _Float16 v8h  __attribute__((ext_vector_type(8)));
typedef _Float16 v16h __attribute__((ext_vector_type(16)));
typedef float    v8f  __attribute__((ext_vector_type(8)));

#define NROWS 8192
#define DIM   256
#define LOG2E 1.4426950408889634f

// ---------------------------------------------------------------- f32 -> f16
__global__ void cvt_f16_kernel(const float* __restrict__ src,
                               _Float16* __restrict__ dst, int n) {
  int i = blockIdx.x * blockDim.x + threadIdx.x;
  if (i < n) dst[i] = (_Float16)src[i];
}

// ------------------------------------------------- h = x @ W^T  (WMMA f16)
__global__ __launch_bounds__(128)
void gemm_h_kernel(const _Float16* __restrict__ x16,
                   const _Float16* __restrict__ w16,
                   float* __restrict__ h) {
  const int lane = threadIdx.x & 31;
  const int wave = threadIdx.x >> 5;
  const int half = lane >> 4;
  const int mr   = lane & 15;
  const int m0   = blockIdx.x * 16;
  const int n0   = wave * 64;

  v8f acc[4] = {};
  for (int k0 = 0; k0 < DIM; k0 += 32) {
    const _Float16* ap = x16 + (size_t)(m0 + mr) * DIM + (k0 + half * 8);
    v8h alo = *(const v8h*)ap;
    v8h ahi = *(const v8h*)(ap + 16);
    v16h a = __builtin_shufflevector(alo, ahi, 0,1,2,3,4,5,6,7,8,9,10,11,12,13,14,15);
#pragma unroll
    for (int nt = 0; nt < 4; ++nt) {
      const _Float16* bp = w16 + (size_t)(n0 + nt * 16 + mr) * DIM + (k0 + half * 8);
      const v16h b = __builtin_shufflevector(*(const v8h*)bp, *(const v8h*)(bp + 16), 0,1,2,3,4,5,6,7,8,9,10,11,12,13,14,15);
      acc[nt] = __builtin_amdgcn_wmma_f32_16x16x32_f16(
          false, a, false, b, (short)0, acc[nt], false, false);
    }
  }
#pragma unroll
  for (int nt = 0; nt < 4; ++nt)
#pragma unroll
    for (int r = 0; r < 8; ++r)
      h[(size_t)(m0 + r + 8 * half) * DIM + (n0 + nt * 16 + mr)] = acc[nt][r];
}

// ------------------------------------------- hT[n, j] = (f16) h[j, n]
__global__ void transpose_f16_kernel(const float* __restrict__ h,
                                     _Float16* __restrict__ hT) {
  __shared__ float tile[32][33];
  int j0 = blockIdx.x * 32, n0 = blockIdx.y * 32;
  int tx = threadIdx.x, ty = threadIdx.y;
#pragma unroll
  for (int r = 0; r < 32; r += 8)
    tile[ty + r][tx] = h[(size_t)(j0 + ty + r) * DIM + (n0 + tx)];
  __syncthreads();
#pragma unroll
  for (int r = 0; r < 32; r += 8)
    hT[(size_t)(n0 + ty + r) * NROWS + (j0 + tx)] = (_Float16)tile[tx][ty + r];
}

// -------- e_i = log2e * (h @ a1), e_j = log2e * (h @ a2)  (log2-domain scores)
__global__ void compute_e_kernel(const float* __restrict__ h,
                                 const float* __restrict__ avec,
                                 float* __restrict__ e_i,
                                 float* __restrict__ e_j) {
  int lane = threadIdx.x & 31, wave = threadIdx.x >> 5;
  int row = blockIdx.x * 8 + wave;
  const float* hr = h + (size_t)row * DIM;
  float s1 = 0.f, s2 = 0.f;
  for (int u = lane; u < DIM; u += 32) {
    float v = hr[u];
    s1 += v * avec[u];
    s2 += v * avec[DIM + u];
  }
#pragma unroll
  for (int off = 16; off > 0; off >>= 1) {
    s1 += __shfl_xor(s1, off, 32);
    s2 += __shfl_xor(s2, off, 32);
  }
  if (lane == 0) { e_i[row] = s1 * LOG2E; e_j[row] = s2 * LOG2E; }
}

// ---------------------------------------------------------------------------
union AM { int4   v[4]; int   e[16]; };
union EV { float4 v[4]; float e[16]; };

// One 32-wide K-block of the flash-softmax loop. Loads the NEXT block into
// (amn, evn), computes from (amc, evc) loaded one step earlier.
__device__ __forceinline__ void attn_step(const int* __restrict__ arow,
                                          const float* __restrict__ e_j,
                                          const _Float16* __restrict__ bb,
                                          int kn,
                                          AM& amc, EV& evc, AM& amn, EV& evn,
                                          float ei, int half,
                                          float& m, float& lsum, v8f* acc) {
  // -- issue next adj / e_j block
  amn.v[0] = *(const int4*)(arow + kn);
  amn.v[1] = *(const int4*)(arow + kn + 4);
  amn.v[2] = *(const int4*)(arow + kn + 16);
  amn.v[3] = *(const int4*)(arow + kn + 20);
  evn.v[0] = *(const float4*)(e_j + kn);
  evn.v[1] = *(const float4*)(e_j + kn + 4);
  evn.v[2] = *(const float4*)(e_j + kn + 16);
  evn.v[3] = *(const float4*)(e_j + kn + 20);

  // -- issue all 8 B-tiles for this block (latency hides under softmax math)
  v16h bt[8];
#pragma unroll
  for (int nt = 0; nt < 8; ++nt)
    bt[nt] = __builtin_shufflevector(*(const v8h*)(bb + (size_t)nt * 16 * NROWS), *(const v8h*)(bb + (size_t)nt * 16 * NROWS + 16), 0,1,2,3,4,5,6,7,8,9,10,11,12,13,14,15);

  // -- block max over e_j (unmasked upper bound; softmax invariant to any m >= max)
  float t[8];
#pragma unroll
  for (int u = 0; u < 8; ++u) t[u] = fmaxf(evc.e[2 * u], evc.e[2 * u + 1]);
#pragma unroll
  for (int u = 0; u < 4; ++u) t[u] = fmaxf(t[u], t[u + 4]);
  t[0] = fmaxf(fmaxf(t[0], t[2]), fmaxf(t[1], t[3]));
  float bm = ei + t[0];
  bm = fmaxf(bm, __shfl_xor(bm, 16, 32));

  const bool  noresc = __all(bm <= m);
  const float mnew   = fmaxf(m, bm);
  const float c      = ei - mnew;          // every exponent (e_j + c) <= 0

  // -- probabilities: p = adj * exp2(e_j + c); masked entries exactly 0
  float ps[4] = {0.f, 0.f, 0.f, 0.f};
  union { v16h v; v2hf f2[8]; } pa;
#pragma unroll
  for (int u = 0; u < 16; u += 2) {
    float p0 = exp2f(evc.e[u] + c)     * (float)amc.e[u];
    float p1 = exp2f(evc.e[u + 1] + c) * (float)amc.e[u + 1];
    ps[(u >> 1) & 3] += p0 + p1;
    pa.f2[u >> 1] = __builtin_amdgcn_cvt_pkrtz(p0, p1);
  }
  float psum = (ps[0] + ps[1]) + (ps[2] + ps[3]);
  psum += __shfl_xor(psum, 16, 32);

  if (noresc) {
    lsum += psum;
  } else {
    const float scale = exp2f(m - mnew);
    lsum = lsum * scale + psum;
    m = mnew;
    float scr[8];
#pragma unroll
    for (int r = 0; r < 8; ++r) scr[r] = __shfl(scale, r + 8 * half, 32);
#pragma unroll
    for (int nt = 0; nt < 8; ++nt)
#pragma unroll
      for (int r = 0; r < 8; ++r) acc[nt][r] *= scr[r];
  }

#pragma unroll
  for (int nt = 0; nt < 8; ++nt)
    acc[nt] = __builtin_amdgcn_wmma_f32_16x16x32_f16(
        false, pa.v, false, bt[nt], (short)0, acc[nt], false, false);
}

// --------------- fused masked-softmax attention + aggregation (flash style)
// block = 128 threads = 4 waves; wave (w>>1) picks 16-row group, (w&1) picks 128-col half.
// K unrolled by 2 with ping-pong buffers (no rotate moves).
__global__ __launch_bounds__(128)
void gat_attn_kernel(const int* __restrict__ adj,
                     const float* __restrict__ e_i,
                     const float* __restrict__ e_j,
                     const _Float16* __restrict__ hT,
                     float* __restrict__ out) {
  const int lane = threadIdx.x & 31;
  const int wave = threadIdx.x >> 5;
  const int half = lane >> 4;
  const int mr   = lane & 15;
  const int i0   = blockIdx.x * 32 + (wave >> 1) * 16;
  const int n0   = (wave & 1) * 128;

  const float ei = e_i[i0 + mr];          // already log2-scaled
  const int*      arow   = adj + (size_t)(i0 + mr) * NROWS;
  const _Float16* bbase0 = hT + (size_t)(n0 + mr) * NROWS + half * 8;

  float m = -1e30f, lsum = 0.f;
  v8f acc[8] = {};

  AM amc, amn;
  EV evc, evn;

  // prologue: load K-block 0
  {
    const int kb = half * 8;
    amc.v[0] = *(const int4*)(arow + kb);
    amc.v[1] = *(const int4*)(arow + kb + 4);
    amc.v[2] = *(const int4*)(arow + kb + 16);
    amc.v[3] = *(const int4*)(arow + kb + 20);
    evc.v[0] = *(const float4*)(e_j + kb);
    evc.v[1] = *(const float4*)(e_j + kb + 4);
    evc.v[2] = *(const float4*)(e_j + kb + 16);
    evc.v[3] = *(const float4*)(e_j + kb + 20);
  }

  for (int k0 = 0; k0 < NROWS; k0 += 64) {
    const int kn1 = (k0 + 32) + half * 8;
    attn_step(arow, e_j, bbase0 + k0, kn1,
              amc, evc, amn, evn, ei, half, m, lsum, acc);
    const int kn2 = ((k0 + 64 < NROWS) ? (k0 + 64) : 0) + half * 8;
    attn_step(arow, e_j, bbase0 + (k0 + 32), kn2,
              amn, evn, amc, evc, ei, half, m, lsum, acc);
  }

  float inv[8];
#pragma unroll
  for (int r = 0; r < 8; ++r) {
    float lr = __shfl(lsum, r + 8 * half, 32);
    inv[r] = 1.0f / lr;
  }
#pragma unroll
  for (int nt = 0; nt < 8; ++nt)
#pragma unroll
    for (int r = 0; r < 8; ++r)
      out[(size_t)(i0 + r + 8 * half) * DIM + (n0 + nt * 16 + mr)] =
          acc[nt][r] * inv[r];
}

// ---------------------------------------------------------------------------
extern "C" void kernel_launch(void* const* d_in, const int* in_sizes, int n_in,
                              void* d_out, int out_size, void* d_ws, size_t ws_size,
                              hipStream_t stream) {
  const float* x   = (const float*)d_in[0];
  const int*   adj = (const int*)  d_in[1];
  const float* W   = (const float*)d_in[2];
  const float* a   = (const float*)d_in[3];
  float* out = (float*)d_out;

  char* ws = (char*)d_ws;
  _Float16* x16 = (_Float16*)(ws);                                  // 4 MB
  _Float16* w16 = (_Float16*)(ws + 4194304);                        // 128 KB
  float*    h   = (float*)   (ws + 4194304 + 131072);               // 8 MB
  _Float16* hT  = (_Float16*)(ws + 4194304 + 131072 + 8388608);     // 4 MB
  float*    e_i = (float*)   (ws + 16908288);                       // 32 KB
  float*    e_j = (float*)   (ws + 16908288 + 32768);               // 32 KB

  cvt_f16_kernel<<<(NROWS * DIM + 255) / 256, 256, 0, stream>>>(x, x16, NROWS * DIM);
  cvt_f16_kernel<<<(DIM * DIM + 255) / 256, 256, 0, stream>>>(W, w16, DIM * DIM);
  gemm_h_kernel<<<NROWS / 16, 128, 0, stream>>>(x16, w16, h);
  transpose_f16_kernel<<<dim3(NROWS / 32, DIM / 32), dim3(32, 8), 0, stream>>>(h, hT);
  compute_e_kernel<<<NROWS / 8, 256, 0, stream>>>(h, a, e_i, e_j);
  gat_attn_kernel<<<NROWS / 32, 128, 0, stream>>>(adj, e_i, e_j, hT, out);
}
